// TextClassification_2637109920219
// MI455X (gfx1250) — matrix operand order repaired and checked by reading, not compiled, not run on hardware
//
#include <hip/hip_runtime.h>
#include <hip/hip_bf16.h>

typedef __attribute__((ext_vector_type(16))) __bf16 bf16x16;
typedef __attribute__((ext_vector_type(8)))  float  f32x8;
typedef __attribute__((ext_vector_type(4)))  unsigned int u32x4;

union FragU {
    bf16x16 bf;
    u32x4   q[2];
};

// ---------------- constants ----------------
#define BTILE 32      // batch rows per workgroup
#define SEQ   80      // timesteps
#define UD    256     // hidden units
#define EPD   128     // embedding dim padded 100 -> 128
#define EDIM  100
#define NB    2048    // batch

// Opaque zero: loop-variant to the optimizer, so weight loads cannot be
// hoisted/persisted across timesteps, but the base pointer stays a kernel
// arg -> address-space inference keeps the loads as global_load (not flat).
__device__ __forceinline__ int opq_zero() {
    int z = 0;
    asm volatile("" : "+s"(z));
    return z;
}

// Hardware tanh (CDNA5 TRANS op). Trailing v_nop covers the TRANS RAW hazard
// the compiler cannot see through the asm block.
__device__ __forceinline__ float htanh(float x) {
    float y;
    asm("v_tanh_f32 %0, %1\n\tv_nop" : "=v"(y) : "v"(x));
    return y;
}

// A-fragment (16x32 bf16, M x K) from row-major bf16 buffer.
// lane: m = lane%16, half = lane/16; v0..3: K = k0+half*8+{0..7}; v4..7: +16
__device__ __forceinline__ bf16x16 loadA(const __bf16* base, int stride,
                                         int row0, int k0, int ln, int lh) {
    const __bf16* p = base + (row0 + ln) * stride + k0 + lh * 8;
    FragU f;
    f.q[0] = *(const u32x4*)(p);
    f.q[1] = *(const u32x4*)(p + 16);
    return f.bf;
}

// B-fragment (32x16 bf16, K x N) from N-major (transposed) weights WT[n][k].
// lane: n = n0+lane%16, half = lane/16; v0..3: K = k0+half*8+{0..7}; v4..7: +16
__device__ __forceinline__ bf16x16 loadB(const __bf16* __restrict__ WT, int Kdim,
                                         int n0, int k0, int ln, int lh) {
    const __bf16* p = WT + (n0 + ln) * Kdim + k0 + lh * 8;
    FragU f;
    f.q[0] = *(const u32x4*)(p);
    f.q[1] = *(const u32x4*)(p + 16);
    return f.bf;
}

// ---------------- prep: fp32 weights -> bf16, transposed to N-major, pad E ----------------
// ws layout (bf16 elems): WxT0[256*128] | WhT0[256*256] | WxT1[256*256] | WhT1[256*256]
__global__ void prep_weights(const float* __restrict__ Wx0,
                             const float* __restrict__ Wx1,
                             const float* __restrict__ Wh,
                             __bf16* __restrict__ ws) {
    int idx = blockIdx.x * blockDim.x + threadIdx.x;
    const int N0 = 256 * 128, N1 = 256 * 256;
    if (idx < N0) {
        int n = idx >> 7, k = idx & 127;
        float v = (k < EDIM) ? Wx0[k * 256 + n] : 0.0f;
        ws[idx] = (__bf16)v;
    } else if (idx < N0 + N1) {
        int i = idx - N0; int n = i >> 8, k = i & 255;
        ws[idx] = (__bf16)Wh[k * 256 + n];                 // Wh[0]
    } else if (idx < N0 + 2 * N1) {
        int i = idx - N0 - N1; int n = i >> 8, k = i & 255;
        ws[idx] = (__bf16)Wx1[k * 256 + n];
    } else if (idx < N0 + 3 * N1) {
        int i = idx - N0 - 2 * N1; int n = i >> 8, k = i & 255;
        ws[idx] = (__bf16)Wh[65536 + k * 256 + n];          // Wh[1]
    }
}

// ---------------- persistent RNN kernel ----------------
__global__ __launch_bounds__(256) void rnn_persistent(
    const int*   __restrict__ tokens,  // [NB, SEQ]
    const float* __restrict__ emb,     // [V, 100]
    const float* __restrict__ bvec,    // [2, 256]
    const float* __restrict__ h0in,    // [2, NB, 256]
    const float* __restrict__ Wo,      // [256]
    const float* __restrict__ bo,      // [1]
    const __bf16* __restrict__ WxT0,   // [256][128] N-major
    const __bf16* __restrict__ WhT0,   // [256][256]
    const __bf16* __restrict__ WxT1,   // [256][256]
    const __bf16* __restrict__ WhT1,   // [256][256]
    float* __restrict__ out)           // [NB]
{
    __shared__ alignas(16) __bf16 sX [BTILE * EPD];
    __shared__ alignas(16) __bf16 sH0[BTILE * UD];
    __shared__ alignas(16) __bf16 sH1[BTILE * UD];

    const int tid   = threadIdx.x;
    const int lane  = tid & 31;
    const int wave  = tid >> 5;         // 0..7
    const int ln    = lane & 15;
    const int lh    = lane >> 4;
    const int bBase = blockIdx.x * BTILE;
    const int nBase = wave * 32;        // each wave owns 32 output columns

    // load initial hidden states into LDS as bf16
    for (int i = tid; i < BTILE * UD; i += 256) {
        int r = i >> 8, u = i & 255;
        sH0[i] = (__bf16)h0in[(bBase + r) * UD + u];
        sH1[i] = (__bf16)h0in[NB * UD + (bBase + r) * UD + u];
    }
    __syncthreads();

    // per-lane bias values (C/D fragment: all 8 elements share column n0+ln)
    const float bias0_0 = bvec[nBase + ln];
    const float bias0_1 = bvec[nBase + 16 + ln];
    const float bias1_0 = bvec[256 + nBase + ln];
    const float bias1_1 = bvec[256 + nBase + 16 + ln];

    for (int t = 0; t < SEQ; ++t) {
        // ---- gather x_t = emb[tokens[:, t]] into LDS (bf16, padded) ----
        for (int i = tid; i < BTILE * EPD; i += 256) {
            int r = i >> 7, k = i & 127;
            int tok = tokens[(bBase + r) * SEQ + t];
            float v = (k < EDIM) ? emb[tok * EDIM + k] : 0.0f;
            sX[i] = (__bf16)v;
        }
        __syncthreads();

        // loop-variant zero offset: keeps weight loads un-hoistable but still
        // recognizably global (global_load_b128 with SGPR base).
        const int z = opq_zero();
        const __bf16* wx0 = WxT0 + z;
        const __bf16* wh0 = WhT0 + z;
        const __bf16* wx1 = WxT1 + z;
        const __bf16* wh1 = WhT1 + z;

        // ================= layer 0: tanh(x@Wx0 + h0@Wh0 + b0) =================
        f32x8 acc[2][2];
        #pragma unroll
        for (int m = 0; m < 2; ++m)
            #pragma unroll
            for (int r = 0; r < 8; ++r) {
                acc[m][0][r] = bias0_0;
                acc[m][1][r] = bias0_1;
            }

        for (int kt = 0; kt < EPD; kt += 32) {           // x @ Wx0
            bf16x16 b0 = loadB(wx0, EPD, nBase,      kt, ln, lh);
            bf16x16 b1 = loadB(wx0, EPD, nBase + 16, kt, ln, lh);
            #pragma unroll
            for (int m = 0; m < 2; ++m) {
                bf16x16 a = loadA(sX, EPD, m * 16, kt, ln, lh);
                acc[m][0] = __builtin_amdgcn_wmma_f32_16x16x32_bf16(false, a, false, b0, (short)0, acc[m][0], false, false);
                acc[m][1] = __builtin_amdgcn_wmma_f32_16x16x32_bf16(false, a, false, b1, (short)0, acc[m][1], false, false);
            }
        }
        for (int kt = 0; kt < UD; kt += 32) {            // h0 @ Wh0
            bf16x16 b0 = loadB(wh0, UD, nBase,      kt, ln, lh);
            bf16x16 b1 = loadB(wh0, UD, nBase + 16, kt, ln, lh);
            #pragma unroll
            for (int m = 0; m < 2; ++m) {
                bf16x16 a = loadA(sH0, UD, m * 16, kt, ln, lh);
                acc[m][0] = __builtin_amdgcn_wmma_f32_16x16x32_bf16(false, a, false, b0, (short)0, acc[m][0], false, false);
                acc[m][1] = __builtin_amdgcn_wmma_f32_16x16x32_bf16(false, a, false, b1, (short)0, acc[m][1], false, false);
            }
        }

        float hv[2][2][8];
        #pragma unroll
        for (int m = 0; m < 2; ++m)
            #pragma unroll
            for (int j = 0; j < 2; ++j)
                #pragma unroll
                for (int r = 0; r < 8; ++r)
                    hv[m][j][r] = htanh(acc[m][j][r]);

        __syncthreads();   // all waves done reading sH0 (old)
        #pragma unroll
        for (int m = 0; m < 2; ++m)
            #pragma unroll
            for (int j = 0; j < 2; ++j)
                #pragma unroll
                for (int r = 0; r < 8; ++r)
                    sH0[(m * 16 + lh * 8 + r) * UD + nBase + j * 16 + ln] = (__bf16)hv[m][j][r];
        __syncthreads();   // sH0 (new) visible to all waves

        // ================= layer 1: tanh(h0@Wx1 + h1@Wh1 + b1) =================
        #pragma unroll
        for (int m = 0; m < 2; ++m)
            #pragma unroll
            for (int r = 0; r < 8; ++r) {
                acc[m][0][r] = bias1_0;
                acc[m][1][r] = bias1_1;
            }

        for (int kt = 0; kt < UD; kt += 32) {            // h0_new @ Wx1
            bf16x16 b0 = loadB(wx1, UD, nBase,      kt, ln, lh);
            bf16x16 b1 = loadB(wx1, UD, nBase + 16, kt, ln, lh);
            #pragma unroll
            for (int m = 0; m < 2; ++m) {
                bf16x16 a = loadA(sH0, UD, m * 16, kt, ln, lh);
                acc[m][0] = __builtin_amdgcn_wmma_f32_16x16x32_bf16(false, a, false, b0, (short)0, acc[m][0], false, false);
                acc[m][1] = __builtin_amdgcn_wmma_f32_16x16x32_bf16(false, a, false, b1, (short)0, acc[m][1], false, false);
            }
        }
        for (int kt = 0; kt < UD; kt += 32) {            // h1 @ Wh1
            bf16x16 b0 = loadB(wh1, UD, nBase,      kt, ln, lh);
            bf16x16 b1 = loadB(wh1, UD, nBase + 16, kt, ln, lh);
            #pragma unroll
            for (int m = 0; m < 2; ++m) {
                bf16x16 a = loadA(sH1, UD, m * 16, kt, ln, lh);
                acc[m][0] = __builtin_amdgcn_wmma_f32_16x16x32_bf16(false, a, false, b0, (short)0, acc[m][0], false, false);
                acc[m][1] = __builtin_amdgcn_wmma_f32_16x16x32_bf16(false, a, false, b1, (short)0, acc[m][1], false, false);
            }
        }

        #pragma unroll
        for (int m = 0; m < 2; ++m)
            #pragma unroll
            for (int j = 0; j < 2; ++j)
                #pragma unroll
                for (int r = 0; r < 8; ++r)
                    hv[m][j][r] = htanh(acc[m][j][r]);

        __syncthreads();   // all waves done reading sH1 (old)
        #pragma unroll
        for (int m = 0; m < 2; ++m)
            #pragma unroll
            for (int j = 0; j < 2; ++j)
                #pragma unroll
                for (int r = 0; r < 8; ++r)
                    sH1[(m * 16 + lh * 8 + r) * UD + nBase + j * 16 + ln] = (__bf16)hv[m][j][r];
        __syncthreads();   // sH1 (new) visible
    }

    // ---- head: sigmoid(h1_final @ Wo + bo) ----
    if (tid < BTILE) {
        float z = bo[0];
        for (int k = 0; k < UD; ++k)
            z += (float)sH1[tid * UD + k] * Wo[k];
        out[bBase + tid] = 1.0f / (1.0f + expf(-z));
    }
}

extern "C" void kernel_launch(void* const* d_in, const int* in_sizes, int n_in,
                              void* d_out, int out_size, void* d_ws, size_t ws_size,
                              hipStream_t stream) {
    const int*   tokens = (const int*)  d_in[0];
    const float* emb    = (const float*)d_in[1];
    const float* Wx0    = (const float*)d_in[2];
    const float* Wx1    = (const float*)d_in[3];
    const float* Wh     = (const float*)d_in[4];
    const float* bvec   = (const float*)d_in[5];
    const float* h0     = (const float*)d_in[6];
    const float* Wo     = (const float*)d_in[7];
    const float* bo     = (const float*)d_in[8];
    float* out = (float*)d_out;

    // workspace: bf16 transposed weights
    unsigned short* wsu = (unsigned short*)d_ws;
    __bf16* wsb  = (__bf16*)d_ws;
    const __bf16* WxT0 = (const __bf16*)(wsu);
    const __bf16* WhT0 = (const __bf16*)(wsu + 256 * 128);
    const __bf16* WxT1 = (const __bf16*)(wsu + 256 * 128 + 256 * 256);
    const __bf16* WhT1 = (const __bf16*)(wsu + 256 * 128 + 2 * 256 * 256);

    const int total = 256 * 128 + 3 * 256 * 256;
    prep_weights<<<(total + 255) / 256, 256, 0, stream>>>(Wx0, Wx1, Wh, wsb);

    rnn_persistent<<<NB / BTILE, 256, 0, stream>>>(
        tokens, emb, bvec, h0, Wo, bo, WxT0, WhT0, WxT1, WhT1, out);
}
